// TopologyAwareModule_24455543783751
// MI455X (gfx1250) — compile-verified, matrix operand order checked by reading, not run on hardware
//
#include <hip/hip_runtime.h>
#include <hip/hip_bf16.h>
#include <stdint.h>

// ---------------------------------------------------------------------------
// Topology-aware loss, MI455X (gfx1250, wave32).
//   K1: bucketize pixels/edges; per-threshold counts via v_wmma_i32_16x16x64_iu8
//       (SWAR byte-compare; sched_barrier-pinned 4-buffer pipeline so the
//        WMMA_IU8->VALU WAR hazard is covered by work, not v_nops)
//   K2: 800x parallel connected-component counting in LDS, bucket plane staged
//       with GLOBAL_LOAD_ASYNC_TO_LDS_B128 (ASYNCcnt)
//   K3: codes -> barcodes -> stable rank sort -> scalar loss
// ---------------------------------------------------------------------------

#define H 128
#define W 128
#define HW (H * W)
#define T_THR 100
#define TPAD 112          // 7 groups of 16 thresholds for WMMA
#define NPLANES 8         // 4 regions x {inside, boundary}
#define NPAIRS 24         // 4 regions x 6 code columns
#define TM1 99            // T-1

typedef int v8i __attribute__((ext_vector_type(8)));

#if __has_builtin(__builtin_amdgcn_sched_barrier)
#define SCHED_FENCE() __builtin_amdgcn_sched_barrier(0)
#else
#define SCHED_FENCE()
#endif

// #{t in [0,99] : p > t/99} == clamp(ceil(99p), 0, 100)   (always <= 100 < 128)
__device__ __forceinline__ int bucketf(float p) {
    float x = 99.0f * p;
    int k = (int)ceilf(x);
    k = k < 0 ? 0 : k;
    k = k > 100 ? 100 : k;
    return k;
}

// Build one 64x16 B-matrix octet of 0x80/0x00 indicator bytes for chunk c.
// SWAR: bytes k,t < 128  =>  ((k + (127-t)) & 0x80) != 0  <=>  k > t; the
// packed 32-bit add cannot carry across byte lanes (max 227).
__device__ __forceinline__ v8i make_b(const uint4* __restrict__ k4,
                                      int c, int grp, unsigned addk) {
    const uint4 lo = k4[c * 4 + grp];         // K 0-31 half
    const uint4 hi = k4[c * 4 + 2 + grp];     // K 32-63 half
    v8i b;
    b[0] = (int)((lo.x + addk) & 0x80808080u);
    b[1] = (int)((lo.y + addk) & 0x80808080u);
    b[2] = (int)((lo.z + addk) & 0x80808080u);
    b[3] = (int)((lo.w + addk) & 0x80808080u);
    b[4] = (int)((hi.x + addk) & 0x80808080u);
    b[5] = (int)((hi.y + addk) & 0x80808080u);
    b[6] = (int)((hi.z + addk) & 0x80808080u);
    b[7] = (int)((hi.w + addk) & 0x80808080u);
    return b;
}

// ---------------------------------------------------------------------------
// Kernel 1: one block per plane. Compute pixel/h-edge/v-edge threshold buckets,
// then per-threshold counts with IU8 WMMA (A = ones 16x64, B = 0x80 indicators,
// D = 128*count, folded out with >>7 at extraction).
// counts layout: counts[(q*8 + plane)*TPAD + t], q in {0:pix, 1:hedge, 2:vedge}
// ---------------------------------------------------------------------------
__global__ void __launch_bounds__(256)
k_bucket_hist(const float* __restrict__ prob,
              const int* __restrict__ sel,
              unsigned char* __restrict__ kbuf,
              int* __restrict__ counts) {
    __shared__ unsigned char s_kp[HW];
    __shared__ unsigned char s_kh[HW];
    __shared__ unsigned char s_kv[HW];

    const int plane = blockIdx.x;         // 0..7
    const int r = plane >> 1;
    const int ch = plane & 1;
    int region = sel[r];
    if (region < 0 || region > 3) region = r;   // defensive clamp
    const float* __restrict__ src = prob + ((size_t)region * 3 + ch) * HW;

    // --- bucketize (single pass; edge bucket = bucket(min(p, neighbor))) ---
    for (int i = threadIdx.x; i < HW; i += 256) {
        __builtin_prefetch(src + i + 2048, 0, 1);   // global_prefetch_b8
        const int row = i >> 7, col = i & 127;
        const float p  = src[i];
        const float pr = (col < (W - 1)) ? src[i + 1] : 0.0f;
        const float pd = (row < (H - 1)) ? src[i + W] : 0.0f;
        const unsigned char kp = (unsigned char)bucketf(p);
        s_kp[i] = kp;
        s_kh[i] = (col < (W - 1)) ? (unsigned char)bucketf(fminf(p, pr)) : 0;
        s_kv[i] = (row < (H - 1)) ? (unsigned char)bucketf(fminf(p, pd)) : 0;
        kbuf[(size_t)plane * HW + i] = kp;          // reused by kernel 2
    }
    __syncthreads();

    // --- WMMA accumulation: 21 tasks (3 quantities x 7 threshold groups) ---
    const int wave = threadIdx.x >> 5;
    const int lane = threadIdx.x & 31;

    v8i a_ones;
#pragma unroll
    for (int k = 0; k < 8; ++k) a_ones[k] = 0x01010101;   // A = all-ones u8 16x64

    for (int task = wave; task < 21; task += 8) {
        const int q = task / 7;        // quantity
        const int g = task % 7;        // threshold group
        const int t0 = g * 16;
        const uint4* __restrict__ k4 =
            (q == 0) ? (const uint4*)s_kp :
            (q == 1) ? (const uint4*)s_kh : (const uint4*)s_kv;

        const unsigned thr  = (unsigned)(t0 + (lane & 15));      // column threshold
        const unsigned addk = (127u - thr) * 0x01010101u;        // SWAR bias
        const int grp = lane >> 4;                               // K sub-block (uint4 units)

        // All four B octets are forced live across the WMMA block by the
        // scheduling fences, so the allocator must use 4 disjoint octets and
        // the next iteration's builds sit >= 3 WMMAs away from their reader.
        v8i acc0 = {}, acc1 = {};
        for (int c = 0; c < HW / 64; c += 4) {                   // 64 iters x 4 chunks
            v8i b0 = make_b(k4, c + 0, grp, addk);
            v8i b1 = make_b(k4, c + 1, grp, addk);
            v8i b2 = make_b(k4, c + 2, grp, addk);
            v8i b3 = make_b(k4, c + 3, grp, addk);
            SCHED_FENCE();
            acc0 = __builtin_amdgcn_wmma_i32_16x16x64_iu8(
                false, a_ones, false, b0, acc0, false, false);
            acc1 = __builtin_amdgcn_wmma_i32_16x16x64_iu8(
                false, a_ones, false, b1, acc1, false, false);
            acc0 = __builtin_amdgcn_wmma_i32_16x16x64_iu8(
                false, a_ones, false, b2, acc0, false, false);
            acc1 = __builtin_amdgcn_wmma_i32_16x16x64_iu8(
                false, a_ones, false, b3, acc1, false, false);
            SCHED_FENCE();
        }
        if (lane < 16) {
            const int t = t0 + lane;
            counts[(q * NPLANES + plane) * TPAD + t] = (acc0[0] + acc1[0]) >> 7;
        }
    }
}

// ---------------------------------------------------------------------------
// Kernel 2: one block per (plane, threshold). 4-connected component count via
// LDS min-label propagation with pointer jumping. b0out[plane*100 + t].
// Bucket plane staged via GLOBAL_LOAD_ASYNC_TO_LDS_B128 when available.
// ---------------------------------------------------------------------------
__global__ void __launch_bounds__(256)
k_components(const unsigned char* __restrict__ kbuf, int* __restrict__ b0out) {
    __shared__ unsigned short lab[HW];     // 32 KB
    __shared__ unsigned char  kb[HW];      // 16 KB
    __shared__ int s_flag;
    __shared__ int s_cnt;

    const int plane = blockIdx.x / T_THR;
    const int t = blockIdx.x % T_THR;
    const unsigned char* __restrict__ src = kbuf + (size_t)plane * HW;

    // stage buckets (16B per lane-transfer)
#if __has_builtin(__builtin_amdgcn_global_load_async_to_lds_b128) && \
    __has_builtin(__builtin_amdgcn_s_wait_asynccnt)
    {
        // builtin prototype (from hipcc diagnostic): params are pointers to
        // gcc-style int4 vectors, global (AS1) src and LDS (AS3) dst.
        typedef int v4i_vs __attribute__((vector_size(16)));
        typedef __attribute__((address_space(1))) v4i_vs gv4i;
        typedef __attribute__((address_space(3))) v4i_vs lv4i;
        for (int i = threadIdx.x; i < HW / 16; i += 256) {
            gv4i* g = (gv4i*)(size_t)(const void*)(src + (size_t)i * 16);
            lv4i* l = (lv4i*)(unsigned)(size_t)(void*)(kb + (size_t)i * 16);
            __builtin_amdgcn_global_load_async_to_lds_b128(g, l, 0, 0);
        }
        __builtin_amdgcn_s_wait_asynccnt(0);
    }
#else
    for (int i = threadIdx.x; i < HW / 16; i += 256)
        ((uint4*)kb)[i] = ((const uint4*)src)[i];
#endif
    __syncthreads();

    const unsigned INACT = 0xFFFFu;
    for (int i = threadIdx.x; i < HW; i += 256)
        lab[i] = (kb[i] > (unsigned char)t) ? (unsigned short)i : (unsigned short)INACT;
    __syncthreads();

    for (;;) {
        if (threadIdx.x == 0) s_flag = 0;
        __syncthreads();
        int changed = 0;
        for (int i = threadIdx.x; i < HW; i += 256) {
            unsigned L = lab[i];
            if (L == INACT) continue;
            const int row = i >> 7, col = i & 127;
            unsigned m = L, v;
            if (col > 0)       { v = lab[i - 1];  m = v < m ? v : m; }
            if (col < W - 1)   { v = lab[i + 1];  m = v < m ? v : m; }
            if (row > 0)       { v = lab[i - W];  m = v < m ? v : m; }
            if (row < H - 1)   { v = lab[i + W];  m = v < m ? v : m; }
            if (m < INACT) {                       // pointer-jump twice
                v = lab[m]; m = v < m ? v : m;
                v = lab[m]; m = v < m ? v : m;
            }
            if (m < L) { lab[i] = (unsigned short)m; changed = 1; }
        }
        if (changed) atomicOr(&s_flag, 1);
        __syncthreads();
        if (!s_flag) break;
    }

    if (threadIdx.x == 0) s_cnt = 0;
    __syncthreads();
    int c = 0;
    for (int i = threadIdx.x; i < HW; i += 256)
        if (lab[i] != INACT && lab[i] == (unsigned short)i) ++c;
    atomicAdd(&s_cnt, c);
    __syncthreads();
    if (threadIdx.x == 0) b0out[plane * T_THR + t] = s_cnt;
}

// ---------------------------------------------------------------------------
// Kernel 3: single block. codes -> barcodes -> stable rank sort -> loss scalar.
// ---------------------------------------------------------------------------
__global__ void __launch_bounds__(256)
k_finalize(const int* __restrict__ counts,
           const int* __restrict__ b0,
           const float* __restrict__ gt,
           float* __restrict__ out) {
    __shared__ float codes[T_THR * NPAIRS];       // 9.6 KB   [t][r*6+j]
    __shared__ float bc[NPAIRS * TM1 * 2];        // 19 KB    (birth, death)
    __shared__ float sgt[NPAIRS * TM1 * 2];       // 19 KB    gt sorted by birth
    __shared__ float gtb[NPAIRS * TM1];           // 9.5 KB   gt births (for ranks)
    __shared__ float s_matched, s_unmatched;

    const int nt = blockDim.x;

    // codes[t][r*6 + {0..5}] = {b0_in, b1_in, b0_bd, b1_bd, b0_in, b1_in}
    for (int idx = threadIdx.x; idx < T_THR * 4; idx += nt) {
        const int t = idx >> 2, r = idx & 3;
        const int pi = r * 2, pb = r * 2 + 1;
        const float b0i = (float)b0[pi * T_THR + t];
        const float b0b = (float)b0[pb * T_THR + t];
        const float eui = (float)(counts[(0 * NPLANES + pi) * TPAD + t]
                                - counts[(1 * NPLANES + pi) * TPAD + t]
                                - counts[(2 * NPLANES + pi) * TPAD + t]);
        const float eub = (float)(counts[(0 * NPLANES + pb) * TPAD + t]
                                - counts[(1 * NPLANES + pb) * TPAD + t]
                                - counts[(2 * NPLANES + pb) * TPAD + t]);
        float* cc = &codes[t * NPAIRS + r * 6];
        cc[0] = b0i; cc[1] = b0i - eui;
        cc[2] = b0b; cc[3] = b0b - eub;
        cc[4] = b0i; cc[5] = b0i - eui;
    }
    // stage gt births
    for (int idx = threadIdx.x; idx < NPAIRS * TM1; idx += nt)
        gtb[idx] = gt[idx * 2];
    __syncthreads();

    // barcodes from code differences
    for (int idx = threadIdx.x; idx < TM1 * NPAIRS; idx += nt) {
        const int t = idx / NPAIRS, c = idx % NPAIRS;
        const float d = codes[(t + 1) * NPAIRS + c] - codes[t * NPAIRS + c];
        const float th = (float)t / 99.0f;
        bc[(c * TM1 + t) * 2 + 0] = (d > 0.0f) ? th : 0.0f;
        bc[(c * TM1 + t) * 2 + 1] = (d < 0.0f) ? th : 0.0f;
    }
    __syncthreads();

    // stable-sort gt by birth (rank scatter; ties broken by index, as argsort)
    for (int idx = threadIdx.x; idx < NPAIRS * TM1; idx += nt) {
        const int pr = idx / TM1, i = idx % TM1;
        const float bi = gtb[pr * TM1 + i];
        int rank = 0;
        for (int k = 0; k < TM1; ++k) {
            const float bk = gtb[pr * TM1 + k];
            rank += (bk < bi) || (bk == bi && k < i);
        }
        sgt[(pr * TM1 + rank) * 2 + 0] = gt[(pr * TM1 + i) * 2 + 0];
        sgt[(pr * TM1 + rank) * 2 + 1] = gt[(pr * TM1 + i) * 2 + 1];
    }
    if (threadIdx.x == 0) { s_matched = 0.0f; s_unmatched = 0.0f; }
    __syncthreads();

    // matched L1 (our barcodes rank-matched against sorted gt) + unmatched length
    float lm = 0.0f, lu = 0.0f;
    for (int idx = threadIdx.x; idx < NPAIRS * TM1; idx += nt) {
        const int pr = idx / TM1, i = idx % TM1;
        const float bi = bc[(pr * TM1 + i) * 2 + 0];
        const float di = bc[(pr * TM1 + i) * 2 + 1];
        int rank = 0;
        for (int k = 0; k < TM1; ++k) {
            const float bk = bc[(pr * TM1 + k) * 2 + 0];
            rank += (bk < bi) || (bk == bi && k < i);
        }
        lm += fabsf(bi - sgt[(pr * TM1 + rank) * 2 + 0])
            + fabsf(di - sgt[(pr * TM1 + rank) * 2 + 1]);
        lu += (di - bi);
    }
    atomicAdd(&s_matched, lm);
    atomicAdd(&s_unmatched, lu);
    __syncthreads();
    if (threadIdx.x == 0)
        out[0] = s_matched / (float)(NPAIRS * TM1) + s_unmatched / (float)NPAIRS;
}

// ---------------------------------------------------------------------------
extern "C" void kernel_launch(void* const* d_in, const int* in_sizes, int n_in,
                              void* d_out, int out_size, void* d_ws, size_t ws_size,
                              hipStream_t stream) {
    const float* prob = (const float*)d_in[0];      // (4,3,128,128) f32
    const int*   sel  = (const int*)d_in[1];        // (4,) int
    const float* gt   = (const float*)d_in[2];      // (4,6,99,2) f32
    float* out = (float*)d_out;

    unsigned char* kbuf = (unsigned char*)d_ws;                       // 8*16384 u8
    int* counts = (int*)((char*)d_ws + (size_t)NPLANES * HW);         // 3*8*112 i32
    int* b0     = counts + 3 * NPLANES * TPAD;                        // 8*100  i32

    k_bucket_hist<<<NPLANES, 256, 0, stream>>>(prob, sel, kbuf, counts);
    k_components<<<NPLANES * T_THR, 256, 0, stream>>>(kbuf, b0);
    k_finalize<<<1, 256, 0, stream>>>(counts, b0, gt, out);
}
